// CustomMoELayer_32564442038660
// MI455X (gfx1250) — compile-verified, matrix operand order checked by reading, not run on hardware
//
#include <hip/hip_runtime.h>
#include <hip/hip_bf16.h>

// ---------------- problem constants ----------------
constexpr int Bq  = 2;
constexpr int Sq  = 2048;
constexpr int Hh  = 1024;   // hidden
constexpr int Ii  = 1024;   // intermediate
constexpr int Ee  = 32;     // experts (== wave32!)
constexpr int Kk  = 2;      // top-k
constexpr int Tt  = Bq * Sq;        // 4096 tokens
constexpr int CAP = 512;

// GEMM tiling
constexpr int TM  = 128;
constexpr int TN  = 64;
constexpr int TK  = 32;
constexpr int LDT = TK + 8;   // padded LDS row stride (ushorts) to spread banks

typedef __attribute__((ext_vector_type(16))) __bf16 v16bf;
typedef __attribute__((ext_vector_type(8)))  float  v8f;

// ---------------- helpers ----------------
__device__ __forceinline__ unsigned short f2bf(float f) {
    unsigned int u = __builtin_bit_cast(unsigned int, f);
    u += 0x7FFFu + ((u >> 16) & 1u);          // round-to-nearest-even
    return (unsigned short)(u >> 16);
}

// LDS byte offset of a __shared__ pointer (generic ptr low 32 bits == LDS addr)
__device__ __forceinline__ unsigned lds_off(const void* p) {
    return (unsigned)(unsigned long long)p;
}

// gfx1250 async DMA: global -> LDS, 16 bytes per lane, tracked by ASYNCcnt.
__device__ __forceinline__ void async_ld16(unsigned lds_addr, unsigned long long gaddr) {
    asm volatile("global_load_async_to_lds_b128 %0, %1, off"
                 :: "v"(lds_addr), "v"(gaddr) : "memory");
}

// Build a 16x32 bf16 A/B fragment from an LDS tile stored [row][k] (k stride 1,
// row stride = ldstride ushorts).  Per ISA 7.12.2: lane = (row%16) + 16*khalf,
// VGPR v holds K pair at (v&4)*4 + (v&3)*2 + 8*khalf.
__device__ __forceinline__ v16bf load_frag(const unsigned short* lds, int row,
                                           int khalf, int ldstride) {
    union { v16bf v; unsigned int u[8]; } f;
    const unsigned short* base = lds + row * ldstride + (khalf << 3);
#pragma unroll
    for (int i = 0; i < 8; ++i) {
        int kb = ((i & 4) << 2) | ((i & 3) << 1);
        f.u[i] = *(const unsigned int*)(base + kb);
    }
    return f.v;
}

__device__ __forceinline__ v8f wmma_bf16(v16bf a, v16bf b, v8f c) {
    return __builtin_amdgcn_wmma_f32_16x16x32_bf16(
        /*neg_a=*/false, a, /*neg_b=*/false, b,
        /*c_mod=*/(short)0, c, /*reuse_a=*/false, /*reuse_b=*/false);
}

// ---------------- 0) zero fill (ws is poisoned) ----------------
__global__ void zero_kernel(uint4* __restrict__ p, size_t n16) {
    size_t i = (size_t)blockIdx.x * blockDim.x + threadIdx.x;
    size_t stride = (size_t)gridDim.x * blockDim.x;
    uint4 z{0u, 0u, 0u, 0u};
    for (; i < n16; i += stride) p[i] = z;
}

// ---------------- 1) router: logits -> softmax -> top2 ----------------
// one wave32 per token; lane == expert id (E == 32)
__global__ void __launch_bounds__(256)
router_kernel(const float* __restrict__ x, const float* __restrict__ wg,
              float* __restrict__ weights_out, int* __restrict__ topi_out,
              int* __restrict__ eid) {
    const int wave = threadIdx.x >> 5;
    const int lane = threadIdx.x & 31;
    const int t = blockIdx.x * 8 + wave;
    if (t >= Tt) return;
    const float* xt = x + (size_t)t * Hh;

    float acc = 0.f;
    for (int h0 = 0; h0 < Hh; h0 += 32) {
        float xv = xt[h0 + lane];
#pragma unroll
        for (int j = 0; j < 32; ++j) {
            float xb = __shfl(xv, j, 32);
            acc += xb * wg[(size_t)(h0 + j) * Ee + lane];
        }
    }
    // wave softmax over 32 experts
    float m = acc;
#pragma unroll
    for (int off = 16; off > 0; off >>= 1) m = fmaxf(m, __shfl_xor(m, off, 32));
    float p = __expf(acc - m);
    float s = p;
#pragma unroll
    for (int off = 16; off > 0; off >>= 1) s += __shfl_xor(s, off, 32);
    p /= s;
    // top-2 (first occurrence on ties, like lax.top_k)
    float m0 = p;
#pragma unroll
    for (int off = 16; off > 0; off >>= 1) m0 = fmaxf(m0, __shfl_xor(m0, off, 32));
    int c0 = (p == m0) ? lane : 1000;
#pragma unroll
    for (int off = 16; off > 0; off >>= 1) c0 = min(c0, __shfl_xor(c0, off, 32));
    float p2 = (lane == c0) ? -1.f : p;
    float m1 = p2;
#pragma unroll
    for (int off = 16; off > 0; off >>= 1) m1 = fmaxf(m1, __shfl_xor(m1, off, 32));
    int c1 = (p2 == m1) ? lane : 1000;
#pragma unroll
    for (int off = 16; off > 0; off >>= 1) c1 = min(c1, __shfl_xor(c1, off, 32));

    if (lane == 0) {
        float sw = m0 + m1;
        weights_out[t * 2 + 0] = m0 / sw;
        weights_out[t * 2 + 1] = m1 / sw;
        topi_out[t * 2 + 0] = c0;
        topi_out[t * 2 + 1] = c1;
        eid[t * 2 + 0] = c0;
        eid[t * 2 + 1] = c1;
    }
}

// ---------------- 2) deterministic rank-within-expert ----------------
__global__ void pos_kernel(const int* __restrict__ eid, int* __restrict__ pos) {
    const int lane = threadIdx.x;   // one lane per expert
    int cnt = 0;
    for (int i = 0; i < Tt * Kk; ++i) {
        int id = eid[i];            // broadcast read
        if (id == lane) pos[i] = cnt++;
    }
}

// ---------------- 3) scatter tokens into bf16 expert buffers ----------------
__global__ void __launch_bounds__(256)
scatter_kernel(const float* __restrict__ x, const int* __restrict__ eid,
               const int* __restrict__ pos, unsigned short* __restrict__ buf) {
    const int a = blockIdx.x;
    const int e = eid[a];
    const int p = pos[a];
    if (p >= CAP) return;
    const int t = a >> 1;
    const float* xt = x + (size_t)t * Hh;
    unsigned short* b = buf + ((size_t)e * CAP + p) * Hh;
    for (int j = threadIdx.x; j < Hh; j += 256) b[j] = f2bf(xt[j]);
}

// ---------------- 4) fused hm = silu(buf@w1) * (buf@w3)  [bf16 out] -------
// Double-buffered LDS; A tile via async global->LDS DMA (ASYNCcnt),
// B tiles via register prefetch + f32->bf16 convert.
__global__ void __launch_bounds__(256)
gemm13_kernel(const unsigned short* __restrict__ buf,
              const float* __restrict__ w1, const float* __restrict__ w3,
              unsigned short* __restrict__ hm) {
    __shared__ unsigned short Als[2][TM * LDT];   // A  [m][k]
    __shared__ unsigned short B1s[2][TN * LDT];   // B1 transposed: [n][k]
    __shared__ unsigned short B3s[2][TN * LDT];   // B3 transposed: [n][k]

    const int e  = blockIdx.z;
    const int m0 = blockIdx.y * TM;
    const int n0 = blockIdx.x * TN;
    const int tid  = threadIdx.x;
    const int lane = tid & 31;
    const int wave = tid >> 5;
    const int wm = wave & 3;      // 4 waves along M  (32 rows each)
    const int wn = wave >> 2;     // 2 waves along N  (32 cols each)

    const unsigned short* bufe = buf + ((size_t)e * CAP + m0) * Hh;
    const float* w1e = w1 + (size_t)e * Hh * Ii + n0;
    const float* w3e = w3 + (size_t)e * Hh * Ii + n0;

    const v8f vzero = {0.f, 0.f, 0.f, 0.f, 0.f, 0.f, 0.f, 0.f};
    v8f acc1[2][2], acc3[2][2];
#pragma unroll
    for (int i = 0; i < 2; ++i)
#pragma unroll
        for (int j = 0; j < 2; ++j) { acc1[i][j] = vzero; acc3[i][j] = vzero; }

    const int khalf = lane >> 4;
    const int mrow  = lane & 15;

    // A staging: each thread owns two 16B chunks (8 bf16) of the 128x32 tile
    const int c = tid & 3;        // chunk within k-row
    const int r = tid >> 2;       // rows 0..63 (+64 for second chunk)
    // B staging: each thread owns 8 strided f32 per matrix
    const int nn  = tid & 63;
    const int kk0 = tid >> 6;     // 0..3, step 4 over k

    float b1r[8], b3r[8];

    auto issueA = [&](int k0, int b) {
        const unsigned short* g = bufe + (size_t)r * Hh + k0 + c * 8;
        async_ld16(lds_off(&Als[b][r * LDT + c * 8]), (unsigned long long)g);
        async_ld16(lds_off(&Als[b][(r + 64) * LDT + c * 8]),
                   (unsigned long long)(g + (size_t)64 * Hh));
    };
    auto loadB = [&](int k0) {
#pragma unroll
        for (int i = 0; i < 8; ++i) {
            const size_t go = (size_t)(k0 + kk0 + i * 4) * Ii + nn;
            b1r[i] = w1e[go];
            b3r[i] = w3e[go];
        }
    };
    auto storeB = [&](int b) {
#pragma unroll
        for (int i = 0; i < 8; ++i) {
            B1s[b][nn * LDT + kk0 + i * 4] = f2bf(b1r[i]);
            B3s[b][nn * LDT + kk0 + i * 4] = f2bf(b3r[i]);
        }
    };

    constexpr int NT = Hh / TK;   // 32 k-steps
    issueA(0, 0);
    loadB(0);

    for (int kt = 0; kt < NT; ++kt) {
        const int b = kt & 1;
        storeB(b);                               // B tile kt -> LDS[b]
        if (kt + 1 < NT) {
            issueA((kt + 1) * TK, b ^ 1);        // async A tile kt+1 in flight
            loadB((kt + 1) * TK);                // B tile kt+1 into regs
            asm volatile("s_wait_asynccnt 0x2" ::: "memory");  // A tile kt done
        } else {
            asm volatile("s_wait_asynccnt 0x0" ::: "memory");
        }
        __syncthreads();

        v16bf a0  = load_frag(&Als[b][0], wm * 32 + mrow,      khalf, LDT);
        v16bf a1  = load_frag(&Als[b][0], wm * 32 + 16 + mrow, khalf, LDT);
        v16bf b10 = load_frag(&B1s[b][0], wn * 32 + mrow,      khalf, LDT);
        v16bf b11 = load_frag(&B1s[b][0], wn * 32 + 16 + mrow, khalf, LDT);
        v16bf b30 = load_frag(&B3s[b][0], wn * 32 + mrow,      khalf, LDT);
        v16bf b31 = load_frag(&B3s[b][0], wn * 32 + 16 + mrow, khalf, LDT);

        acc1[0][0] = wmma_bf16(a0, b10, acc1[0][0]);
        acc1[0][1] = wmma_bf16(a0, b11, acc1[0][1]);
        acc1[1][0] = wmma_bf16(a1, b10, acc1[1][0]);
        acc1[1][1] = wmma_bf16(a1, b11, acc1[1][1]);
        acc3[0][0] = wmma_bf16(a0, b30, acc3[0][0]);
        acc3[0][1] = wmma_bf16(a0, b31, acc3[0][1]);
        acc3[1][0] = wmma_bf16(a1, b30, acc3[1][0]);
        acc3[1][1] = wmma_bf16(a1, b31, acc3[1][1]);
        __syncthreads();
    }

    // epilogue: silu(h1)*h3 -> bf16 hm
    unsigned short* hme = hm + (size_t)e * CAP * Ii;
    const int col = lane & 15;
    const int mh  = lane >> 4;
#pragma unroll
    for (int mt = 0; mt < 2; ++mt)
#pragma unroll
        for (int nt = 0; nt < 2; ++nt) {
            const int gn = n0 + wn * 32 + nt * 16 + col;
#pragma unroll
            for (int v = 0; v < 8; ++v) {
                const int gm = m0 + wm * 32 + mt * 16 + v + 8 * mh;
                float h1 = acc1[mt][nt][v];
                float h3 = acc3[mt][nt][v];
                float sv = h1 / (1.f + __expf(-h1));
                hme[(size_t)gm * Ii + gn] = f2bf(sv * h3);
            }
        }
}

// ---------------- 5) out_e = hm @ w2   [f32 out] ----------------
__global__ void __launch_bounds__(256)
gemm2_kernel(const unsigned short* __restrict__ hm,
             const float* __restrict__ w2, float* __restrict__ oute) {
    __shared__ unsigned short Als[2][TM * LDT];
    __shared__ unsigned short Bs[2][TN * LDT];

    const int e  = blockIdx.z;
    const int m0 = blockIdx.y * TM;
    const int n0 = blockIdx.x * TN;
    const int tid  = threadIdx.x;
    const int lane = tid & 31;
    const int wave = tid >> 5;
    const int wm = wave & 3;
    const int wn = wave >> 2;

    const unsigned short* hme = hm + ((size_t)e * CAP + m0) * Ii;
    const float* w2e = w2 + (size_t)e * Ii * Hh + n0;

    const v8f vzero = {0.f, 0.f, 0.f, 0.f, 0.f, 0.f, 0.f, 0.f};
    v8f acc[2][2];
#pragma unroll
    for (int i = 0; i < 2; ++i)
#pragma unroll
        for (int j = 0; j < 2; ++j) acc[i][j] = vzero;

    const int khalf = lane >> 4;
    const int mrow  = lane & 15;

    const int c = tid & 3;
    const int r = tid >> 2;
    const int nn  = tid & 63;
    const int kk0 = tid >> 6;

    float br[8];

    auto issueA = [&](int k0, int b) {
        const unsigned short* g = hme + (size_t)r * Ii + k0 + c * 8;
        async_ld16(lds_off(&Als[b][r * LDT + c * 8]), (unsigned long long)g);
        async_ld16(lds_off(&Als[b][(r + 64) * LDT + c * 8]),
                   (unsigned long long)(g + (size_t)64 * Ii));
    };
    auto loadB = [&](int k0) {
#pragma unroll
        for (int i = 0; i < 8; ++i)
            br[i] = w2e[(size_t)(k0 + kk0 + i * 4) * Hh + nn];
    };
    auto storeB = [&](int b) {
#pragma unroll
        for (int i = 0; i < 8; ++i)
            Bs[b][nn * LDT + kk0 + i * 4] = f2bf(br[i]);
    };

    constexpr int NT = Ii / TK;
    issueA(0, 0);
    loadB(0);

    for (int kt = 0; kt < NT; ++kt) {
        const int b = kt & 1;
        storeB(b);
        if (kt + 1 < NT) {
            issueA((kt + 1) * TK, b ^ 1);
            loadB((kt + 1) * TK);
            asm volatile("s_wait_asynccnt 0x2" ::: "memory");
        } else {
            asm volatile("s_wait_asynccnt 0x0" ::: "memory");
        }
        __syncthreads();

        v16bf a0 = load_frag(&Als[b][0], wm * 32 + mrow,      khalf, LDT);
        v16bf a1 = load_frag(&Als[b][0], wm * 32 + 16 + mrow, khalf, LDT);
        v16bf b0 = load_frag(&Bs[b][0],  wn * 32 + mrow,      khalf, LDT);
        v16bf b1 = load_frag(&Bs[b][0],  wn * 32 + 16 + mrow, khalf, LDT);

        acc[0][0] = wmma_bf16(a0, b0, acc[0][0]);
        acc[0][1] = wmma_bf16(a0, b1, acc[0][1]);
        acc[1][0] = wmma_bf16(a1, b0, acc[1][0]);
        acc[1][1] = wmma_bf16(a1, b1, acc[1][1]);
        __syncthreads();
    }

    float* oe = oute + (size_t)e * CAP * Hh;
    const int col = lane & 15;
    const int mh  = lane >> 4;
#pragma unroll
    for (int mt = 0; mt < 2; ++mt)
#pragma unroll
        for (int nt = 0; nt < 2; ++nt) {
            const int gn = n0 + wn * 32 + nt * 16 + col;
#pragma unroll
            for (int v = 0; v < 8; ++v) {
                const int gm = m0 + wm * 32 + mt * 16 + v + 8 * mh;
                oe[(size_t)gm * Hh + gn] = acc[mt][nt][v];
            }
        }
}

// ---------------- 6) combine: weighted sum + perf norms ----------------
__global__ void __launch_bounds__(256)
combine_kernel(const float* __restrict__ oute, const int* __restrict__ eid,
               const int* __restrict__ pos, const float* __restrict__ weights,
               float* __restrict__ final_out, float* __restrict__ perf) {
    __shared__ float red[8];
    const int t = blockIdx.x;
    const int tid = threadIdx.x;
    float f[4] = {0.f, 0.f, 0.f, 0.f};

    for (int k = 0; k < Kk; ++k) {
        const int a = t * Kk + k;
        const int e = eid[a];
        const int p = pos[a];
        const float w = weights[a];
        float ss = 0.f;
        if (p < CAP) {
            const float* y = oute + ((size_t)e * CAP + p) * Hh;
#pragma unroll
            for (int j = 0; j < 4; ++j) {
                float v = y[tid + j * 256];
                f[j] += w * v;
                ss += v * v;
            }
        }
#pragma unroll
        for (int off = 16; off > 0; off >>= 1) ss += __shfl_xor(ss, off, 32);
        if ((tid & 31) == 0) red[tid >> 5] = ss;
        __syncthreads();
        if (tid < 8) {
            float r = red[tid];
#pragma unroll
            for (int off = 4; off > 0; off >>= 1) r += __shfl_xor(r, off, 32);
            if (tid == 0) perf[a] = sqrtf(r);
        }
        __syncthreads();
    }

    float* ft = final_out + (size_t)t * Hh;
#pragma unroll
    for (int j = 0; j < 4; ++j) ft[tid + j * 256] = f[j];
}

// ---------------- host launcher ----------------
extern "C" void kernel_launch(void* const* d_in, const int* in_sizes, int n_in,
                              void* d_out, int out_size, void* d_ws, size_t ws_size,
                              hipStream_t stream) {
    const float* x  = (const float*)d_in[0];
    const float* wg = (const float*)d_in[1];
    const float* w1 = (const float*)d_in[2];
    const float* w2 = (const float*)d_in[3];
    const float* w3 = (const float*)d_in[4];
    float* out = (float*)d_out;

    // output layout (flat, in return order): final | weights | topi | perf
    const size_t OFF_W = (size_t)Tt * Hh;              // 4194304
    const size_t OFF_I = OFF_W + (size_t)Tt * Kk;      // + 8192
    const size_t OFF_P = OFF_I + (size_t)Tt * Kk;      // + 8192

    // workspace layout
    char* ws = (char*)d_ws;
    const size_t BUF_BYTES = (size_t)Ee * CAP * Hh * 2;   // 32 MB bf16
    const size_t HM_BYTES  = (size_t)Ee * CAP * Ii * 2;   // 32 MB bf16
    const size_t OUTE_BYTES= (size_t)Ee * CAP * Hh * 4;   // 64 MB f32
    unsigned short* buf  = (unsigned short*)ws;
    unsigned short* hm   = (unsigned short*)(ws + BUF_BYTES);
    float*          oute = (float*)(ws + BUF_BYTES + HM_BYTES);
    int* eid = (int*)(ws + BUF_BYTES + HM_BYTES + OUTE_BYTES);
    int* pos = eid + Tt * Kk;

    // 0) zero the expert input buffers (ws is poisoned)
    zero_kernel<<<4096, 256, 0, stream>>>((uint4*)buf, BUF_BYTES / 16);
    // 1) router
    router_kernel<<<Tt / 8, 256, 0, stream>>>(x, wg, out + OFF_W, (int*)(out + OFF_I), eid);
    // 2) deterministic rank within expert
    pos_kernel<<<1, 32, 0, stream>>>(eid, pos);
    // 3) scatter tokens -> bf16 expert buffers
    scatter_kernel<<<Tt * Kk, 256, 0, stream>>>(x, eid, pos, buf);
    // 4) fused w1/w3 GEMM + SwiGLU
    gemm13_kernel<<<dim3(Ii / TN, CAP / TM, Ee), 256, 0, stream>>>(buf, w1, w3, hm);
    // 5) w2 GEMM
    gemm2_kernel<<<dim3(Hh / TN, CAP / TM, Ee), 256, 0, stream>>>(hm, w2, oute);
    // 6) combine + perf
    combine_kernel<<<Tt, 256, 0, stream>>>(oute, eid, pos, out + OFF_W, out, out + OFF_P);
}
